// GravnetModel_4501125726944
// MI455X (gfx1250) — matrix operand order repaired and checked by reading, not compiled, not run on hardware
//
#include <hip/hip_runtime.h>
#include <hip/hip_bf16.h>

// GravNet forward for MI455X (gfx1250, wave32, WMMA).
// All 42 linear layers -> one generic WMMA f16 GEMM kernel (f32 in/out,
// f16 LDS staging, v_wmma_f32_16x16x32_f16). kNN + aggregation and the
// per-event global-exchange reductions are VALU kernels (L2-resident data).
// kNN coordinate staging uses gfx1250 async global->LDS copies when the
// builtins are available (guarded; falls back to plain LDS stores).

typedef __attribute__((ext_vector_type(16))) _Float16 v16h;
typedef __attribute__((ext_vector_type(8)))  float    v8f;

#define NTOT     65536   // B * N_PER
#define NEV      64
#define NPER     1024
#define TILE_R   128     // rows per workgroup in GEMM
#define MAX_CIN  160     // padded-K upper bound (lin_out: 32+128)
#define MAX_COUT 64
#define KNN      7

#if defined(__has_builtin)
#if __has_builtin(__builtin_amdgcn_global_load_async_to_lds_b32) && \
    __has_builtin(__builtin_amdgcn_s_wait_asynccnt)
#define USE_ASYNC_LDS 1
#endif
#endif

typedef __attribute__((address_space(1))) int gint_t;
typedef __attribute__((address_space(3))) int lint_t;

// ---------------------------------------------------------------------------
// Generic Y = act(X @ W^T + b).  X: nrows x cin (row stride ldx, f32)
// W: cout x cin row-major (f32).  Y: row stride ldy (f32).
// act: 0 = identity, 1 = ELU.
// 256 threads = 8 waves; wave w computes rows [blk*128 + 16w, +16).
// ---------------------------------------------------------------------------
__launch_bounds__(256)
__global__ void wmma_linear(const float* __restrict__ X, int ldx,
                            const float* __restrict__ W,
                            const float* __restrict__ bias,
                            float* __restrict__ Y, int ldy,
                            int nrows, int cin, int cout, int act)
{
    __shared__ _Float16 sX[TILE_R * MAX_CIN];
    __shared__ _Float16 sW[MAX_COUT * MAX_CIN];

    const int cinp  = (cin  + 31) & ~31;   // K padded to 32
    const int coutp = (cout + 15) & ~15;   // N padded to 16
    const int tid     = threadIdx.x;
    const int rowbase = blockIdx.x * TILE_R;

    // Stage X tile as f16 (zero-padded K).
    for (int idx = tid; idx < TILE_R * cinp; idx += 256) {
        const int r = idx / cinp, c = idx - r * cinp;
        const int row = rowbase + r;
        float v = (row < nrows && c < cin) ? X[(size_t)row * ldx + c] : 0.0f;
        sX[r * cinp + c] = (_Float16)v;
    }
    // Stage W as f16 (zero-padded K and N).
    for (int idx = tid; idx < coutp * cinp; idx += 256) {
        const int r = idx / cinp, c = idx - r * cinp;
        float v = (r < cout && c < cin) ? W[(size_t)r * cin + c] : 0.0f;
        sW[r * cinp + c] = (_Float16)v;
    }
    __syncthreads();

    const int wave = tid >> 5;
    const int lane = tid & 31;
    const int g    = lane >> 4;      // lane-half
    const int m    = lane & 15;

    // A fragment source row (A: 16x32, lane holds row m; K halves split by g).
    const _Float16* arow = &sX[(wave * 16 + m) * cinp];

    for (int n0 = 0; n0 < coutp; n0 += 16) {
        v8f acc = {};
        const _Float16* brow = &sW[(n0 + m) * cinp];  // B col = lane&15
        for (int k0 = 0; k0 < cinp; k0 += 32) {
            v16h a, b;
            // A 16x32 f16 layout: lanes0-15 K {0..7,16..23}, lanes16-31 K {8..15,24..31}
            #pragma unroll
            for (int e = 0; e < 8; ++e) {
                a[e]     = arow[k0 + 8 * g + e];
                a[e + 8] = arow[k0 + 16 + 8 * g + e];
            }
            // B 32x16 f16 layout: lanes0-15 K {0..7,16..23}, lanes16-31 K {8..15,24..31}
            #pragma unroll
            for (int e = 0; e < 8; ++e) {
                b[e]     = brow[k0 + 8 * g + e];
                b[e + 8] = brow[k0 + 16 + 8 * g + e];
            }

            acc = __builtin_amdgcn_wmma_f32_16x16x32_f16(
                      false, a, false, b, (short)0, acc, false, false);
        }
        const int col = n0 + m;
        const float bv = (bias != nullptr && col < cout) ? bias[col] : 0.0f;
        #pragma unroll
        for (int r = 0; r < 8; ++r) {
            const int row = rowbase + wave * 16 + r + 8 * g;  // C/D: M = r + 8*g
            if (row < nrows && col < cout) {
                float v = acc[r] + bv;
                if (act) v = v > 0.0f ? v : (__expf(v) - 1.0f);
                Y[(size_t)row * ldy + col] = v;
            }
        }
    }
}

// ---------------------------------------------------------------------------
// Per-event kNN (K=7, 3-dim learned space) + distance-weighted mean/max
// aggregation of propagated features.  Writes concat row [x(32)|mean(64)|max(64)].
// Grid: 4 workgroups per event, 256 threads, one node per thread.
// ---------------------------------------------------------------------------
__launch_bounds__(256)
__global__ void knn_aggregate(const float* __restrict__ S, int sld,
                              const float* __restrict__ H,     // N x 64
                              const float* __restrict__ Xin,   // N x 32
                              float* __restrict__ XA)          // N x 160
{
    __shared__ float ss[NPER * 3];
    const int e    = blockIdx.x >> 2;
    const int tid  = threadIdx.x;
    const int base = e * NPER;

#if defined(USE_ASYNC_LDS) && defined(__gfx1250__)
    // gfx1250 async global->LDS staging (ASYNCcnt path). Rows are only 4B
    // aligned (stride 35 floats) so issue 3x b32; the instruction offset is
    // added to BOTH addresses, matching the contiguous ss[j*3+d] layout.
    for (int j = tid; j < NPER; j += 256) {
        const float* sp = S + (size_t)(base + j) * sld;
        gint_t* gp = (gint_t*)(void*)sp;
        lint_t* lp = (lint_t*)&ss[j * 3];
        __builtin_amdgcn_global_load_async_to_lds_b32(gp, lp, 0, 0);
        __builtin_amdgcn_global_load_async_to_lds_b32(gp, lp, 4, 0);
        __builtin_amdgcn_global_load_async_to_lds_b32(gp, lp, 8, 0);
    }
    __builtin_amdgcn_s_wait_asynccnt(0);
#else
    for (int j = tid; j < NPER; j += 256) {
        const float* sp = S + (size_t)(base + j) * sld;
        ss[j * 3 + 0] = sp[0];
        ss[j * 3 + 1] = sp[1];
        ss[j * 3 + 2] = sp[2];
    }
#endif
    __syncthreads();

    const int i = (blockIdx.x & 3) * 256 + tid;   // node within event
    const float sx = ss[i * 3 + 0], sy = ss[i * 3 + 1], sz = ss[i * 3 + 2];

    float bd[KNN]; int bj[KNN];
    #pragma unroll
    for (int k = 0; k < KNN; ++k) { bd[k] = 3.4e38f; bj[k] = 0; }

    for (int j = 0; j < NPER; ++j) {
        const float dx = sx - ss[j * 3 + 0];
        const float dy = sy - ss[j * 3 + 1];
        const float dz = sz - ss[j * 3 + 2];
        const float d2 = fmaf(dx, dx, fmaf(dy, dy, dz * dz));
        if (d2 < bd[KNN - 1]) {          // replace worst, bubble into place
            bd[KNN - 1] = d2; bj[KNN - 1] = j;
            #pragma unroll
            for (int k = KNN - 1; k > 0; --k) {
                if (bd[k] < bd[k - 1]) {
                    float td = bd[k]; bd[k] = bd[k - 1]; bd[k - 1] = td;
                    int   tj = bj[k]; bj[k] = bj[k - 1]; bj[k - 1] = tj;
                }
            }
        }
    }

    float w[KNN];
    const float* hrow[KNN];
    #pragma unroll
    for (int k = 0; k < KNN; ++k) {
        w[k] = __expf(-10.0f * bd[k]);
        hrow[k] = H + (size_t)(base + bj[k]) * 64;
        __builtin_prefetch(hrow[k], 0, 0);     // global_prefetch_b8
    }

    float* orow = XA + (size_t)(base + i) * 160;
    const float* xr = Xin + (size_t)(base + i) * 32;
    #pragma unroll
    for (int c = 0; c < 32; ++c) orow[c] = xr[c];

    for (int f0 = 0; f0 < 64; f0 += 16) {
        float mean[16], mx[16];
        #pragma unroll
        for (int t = 0; t < 16; ++t) { mean[t] = 0.0f; mx[t] = -3.4e38f; }
        #pragma unroll
        for (int k = 0; k < KNN; ++k) {
            #pragma unroll
            for (int t = 0; t < 16; ++t) {
                const float v = hrow[k][f0 + t] * w[k];
                mean[t] += v;
                mx[t] = fmaxf(mx[t], v);
            }
        }
        #pragma unroll
        for (int t = 0; t < 16; ++t) {
            orow[32 + f0 + t] = mean[t] * (1.0f / (float)KNN);
            orow[96 + f0 + t] = mx[t];
        }
    }
}

// ---------------------------------------------------------------------------
// Per-event global exchange: mean/min/max over the 1024 nodes of each event,
// broadcast as rows [mean(32)|min(32)|max(32)|x(32)].  One workgroup per event.
// ---------------------------------------------------------------------------
__launch_bounds__(256)
__global__ void global_exchange_k(const float* __restrict__ X,  // N x 32
                                  float* __restrict__ GE)       // N x 128
{
    __shared__ float ssum[8][32], smin[8][32], smax[8][32];
    __shared__ float gmean[32], gmin[32], gmax[32];
    const int e   = blockIdx.x;
    const int tid = threadIdx.x;
    const int d   = tid & 31, sl = tid >> 5;

    float s = 0.0f, mn = 3.4e38f, mx = -3.4e38f;
    for (int j = sl; j < NPER; j += 8) {
        const float v = X[((size_t)e * NPER + j) * 32 + d];
        s += v; mn = fminf(mn, v); mx = fmaxf(mx, v);
    }
    ssum[sl][d] = s; smin[sl][d] = mn; smax[sl][d] = mx;
    __syncthreads();
    if (sl == 0) {
        #pragma unroll
        for (int k = 1; k < 8; ++k) {
            s += ssum[k][d];
            mn = fminf(mn, smin[k][d]);
            mx = fmaxf(mx, smax[k][d]);
        }
        gmean[d] = s * (1.0f / (float)NPER);
        gmin[d] = mn; gmax[d] = mx;
    }
    __syncthreads();

    for (int idx = tid; idx < NPER * 128; idx += 256) {
        const int row = idx >> 7, col = idx & 127;
        float v;
        if      (col < 32) v = gmean[col];
        else if (col < 64) v = gmin[col - 32];
        else if (col < 96) v = gmax[col - 64];
        else               v = X[((size_t)e * NPER + row) * 32 + (col - 96)];
        GE[((size_t)e * NPER + row) * 128 + col] = v;
    }
}

// ---------------------------------------------------------------------------
// Orchestration
// ---------------------------------------------------------------------------
static inline void lin(const float* X, int ldx, const void* W, const void* b,
                       float* Y, int ldy, int cin, int cout, int act,
                       hipStream_t st)
{
    dim3 grid((NTOT + TILE_R - 1) / TILE_R);
    wmma_linear<<<grid, 256, 0, st>>>(X, ldx, (const float*)W, (const float*)b,
                                      Y, ldy, NTOT, cin, cout, act);
}

extern "C" void kernel_launch(void* const* d_in, const int* in_sizes, int n_in,
                              void* d_out, int out_size, void* d_ws, size_t ws_size,
                              hipStream_t stream)
{
    (void)in_sizes; (void)n_in; (void)out_size; (void)ws_size;
    const size_t N = NTOT;

    // Input flattening (insertion order of setup_inputs / params dicts):
    //  [0]=x (N,9)  [1]=batch (unused)  [2]=w_in (64,9)
    //  [3 + i*16 + j*2 (+1)]: block i, layer j in
    //      {pre1,pre2,lin_s,lin_h,lin_out,post1,post2,out}, (w,b)
    //  [67..74]=dense 4x(w,b)  [75..80]=outstack 3x(w,b)
    //  [81,82]=clust (w,b)     [83,84]=beta (w,b)
    const float* x_in = (const float*)d_in[0];
    const float* w_in = (const float*)d_in[2];
    auto BW = [&](int i, int j) { return d_in[3 + i * 16 + j * 2]; };
    auto BB = [&](int i, int j) { return d_in[3 + i * 16 + j * 2 + 1]; };

    // Workspace layout (floats)
    float* ws    = (float*)d_ws;
    float* h0    = ws;               // N x 64   input-linear output
    float* t1    = h0    +  64 * N;  // N x 32
    float* t2    = t1    +  32 * N;  // N x 32   conv input x
    float* cat35 = t2    +  32 * N;  // N x 35   [lin_out(32) | s(3)]
    float* hp    = cat35 +  35 * N;  // N x 64   propagated features
    float* big   = hp    +  64 * N;  // N x 160  xa; reused as ge (N x 128)
    float* p2    = big   + 160 * N;  // N x 32   post2 output
    float* feats = p2    +  32 * N;  // N x 128  block outputs (col 32*i)
    float* z1    = feats + 128 * N;  // N x 64
    float* z2    = z1    +  64 * N;  // N x 64

    // Input linear (bias=False, no act): h0 = x @ w_in^T
    lin(x_in, 9, w_in, nullptr, h0, 64, 9, 64, 0, stream);

    for (int i = 0; i < 4; ++i) {
        const float* in  = (i == 0) ? h0 : (feats + 32 * (i - 1));
        const int ldin   = (i == 0) ? 64 : 128;
        const int cin0   = (i == 0) ? 64 : 32;

        lin(in, ldin, BW(i,0), BB(i,0), t1, 32, cin0, 32, 1, stream);   // pre1+ELU
        lin(t1, 32,   BW(i,1), BB(i,1), t2, 32, 32,   32, 1, stream);   // pre2+ELU
        lin(t2, 32,   BW(i,2), BB(i,2), cat35 + 32, 35, 32, 3, 0, stream); // lin_s
        lin(t2, 32,   BW(i,3), BB(i,3), hp, 64, 32,  64, 0, stream);    // lin_h

        knn_aggregate<<<NEV * 4, 256, 0, stream>>>(cat35 + 32, 35, hp, t2, big);

        lin(big, 160, BW(i,4), BB(i,4), cat35, 35, 160, 32, 0, stream); // lin_out
        lin(cat35, 35, BW(i,5), BB(i,5), t1, 32, 35, 32, 1, stream);    // post1+ELU
        lin(t1, 32,   BW(i,6), BB(i,6), p2, 32, 32, 32, 1, stream);     // post2+ELU

        global_exchange_k<<<NEV, 256, 0, stream>>>(p2, big);

        lin(big, 128, BW(i,7), BB(i,7), feats + 32 * i, 128, 128, 32, 1, stream);
    }

    // Dense stack (4x ELU)
    lin(feats, 128, d_in[67], d_in[68], z1, 64, 128, 64, 1, stream);
    lin(z1, 64,     d_in[69], d_in[70], z2, 64, 64,  64, 1, stream);
    lin(z2, 64,     d_in[71], d_in[72], z1, 64, 64,  64, 1, stream);
    lin(z1, 64,     d_in[73], d_in[74], z2, 64, 64,  64, 1, stream);
    // Output stack: ELU, ELU, linear
    lin(z2, 64,     d_in[75], d_in[76], z1, 64, 64,  64, 1, stream);
    lin(z1, 64,     d_in[77], d_in[78], z2, 64, 64,  64, 1, stream);
    lin(z2, 64,     d_in[79], d_in[80], z1, 64, 64,  64, 0, stream);
    // Heads into d_out (N x 31): clust -> cols [0,30), beta -> col 30
    float* out = (float*)d_out;
    lin(z1, 64, d_in[81], d_in[82], out,      31, 64, 30, 0, stream);
    lin(z1, 64, d_in[83], d_in[84], out + 30, 31, 64,  1, 0, stream);
}
